// W8A8Linear_927712936111
// MI455X (gfx1250) — compile-verified, matrix operand order checked by reading, not run on hardware
//
#include <hip/hip_runtime.h>

// W8A8 linear: per-token dynamic int8 quant -> int8 GEMM (WMMA IU8) -> fused dequant+bias.
// M=8192 tokens, K=4096, N=12288. Arithmetic intensity ~2700 op/byte => compute-bound;
// use V_WMMA_I32_16X16X64_IU8 with an async (memory->LDS DMA) pipeline.

typedef __attribute__((ext_vector_type(8))) int      v8i;
typedef __attribute__((ext_vector_type(8))) _Float16 v8h;

#define M_TOK 8192
#define K_DIM 4096
#define N_DIM 12288

#define BM 64     // block tile M (2 waves * 32)
#define BN 128    // block tile N (4 waves * 32)
#define BK 128    // K bytes staged per LDS stage (2 WMMA k-steps of 64)
#define LDSB 144  // padded LDS row stride (bytes): 16B-aligned, avoids bank conflicts

// ---------------- CDNA5 async memory->LDS copy (ASYNCcnt-tracked) ----------------
__device__ __forceinline__ void async_copy_b128(unsigned lds_addr, const void* gptr) {
  // GLOBAL_LOAD_ASYNC_TO_LDS_B128: VDST = LDS byte address VGPR, VADDR = 64-bit global addr.
  asm volatile("global_load_async_to_lds_b128 %0, %1, off"
               :
               : "v"(lds_addr), "v"(gptr)
               : "memory");
}
__device__ __forceinline__ void wait_async0() {
  asm volatile("s_wait_asynccnt 0x0" ::: "memory");
}

// ---------------- Kernel 1: per-token activation quantization ----------------
__global__ __launch_bounds__(256) void quant_x_kernel(const _Float16* __restrict__ x,
                                                      signed char* __restrict__ xq,
                                                      float* __restrict__ xs) {
  __shared__ float red[256];
  const int row = blockIdx.x;
  const int t = threadIdx.x;
  const _Float16* xr = x + (size_t)row * K_DIM;

  v8h a = *(const v8h*)(xr + t * 16);
  v8h b = *(const v8h*)(xr + t * 16 + 8);
  float v[16];
#pragma unroll
  for (int i = 0; i < 8; ++i) { v[i] = (float)a[i]; v[8 + i] = (float)b[i]; }
  float amax = 0.0f;
#pragma unroll
  for (int i = 0; i < 16; ++i) amax = fmaxf(amax, fabsf(v[i]));
  red[t] = amax;
  __syncthreads();
  for (int s = 128; s > 0; s >>= 1) {
    if (t < s) red[t] = fmaxf(red[t], red[t + s]);
    __syncthreads();
  }
  const float scale = fmaxf(red[0], 1e-6f) * (1.0f / 127.0f);
  const float inv = 1.0f / scale;

  int packed[4];
#pragma unroll
  for (int g = 0; g < 4; ++g) {
    int q[4];
#pragma unroll
    for (int e = 0; e < 4; ++e) {
      int qi = __float2int_rn(v[g * 4 + e] * inv);  // round-to-nearest-even like jnp.round
      qi = qi < -128 ? -128 : (qi > 127 ? 127 : qi);
      q[e] = qi & 0xFF;
    }
    packed[g] = q[0] | (q[1] << 8) | (q[2] << 16) | (q[3] << 24);
  }
  *(int4*)(xq + (size_t)row * K_DIM + t * 16) = make_int4(packed[0], packed[1], packed[2], packed[3]);
  if (t == 0) xs[row] = scale;
}

// ---------------- Kernel 2: pack W int32 -> int8 (harness passes integers as int32) ----
__global__ __launch_bounds__(256) void pack_w_kernel(const int* __restrict__ w,
                                                     unsigned int* __restrict__ wq) {
  const size_t i = (size_t)blockIdx.x * 256 + threadIdx.x;  // one packed dword (4 weights)
  int4 v = ((const int4*)w)[i];
  wq[i] = (unsigned)(v.x & 0xFF) | ((unsigned)(v.y & 0xFF) << 8) |
          ((unsigned)(v.z & 0xFF) << 16) | ((unsigned)(v.w & 0xFF) << 24);
}

// ---------------- Fragment load: 8-bit A/B 16x64 layout ----------------
// Per ISA: lanes 0-15 hold K-groups {0-7,16-23,32-39,48-55}, lanes 16-31 the +8 halves.
// Each fragment = 4 conflict-free ds_load_b64 (compiler pairs them into ds_load_2addr_b64).
__device__ __forceinline__ v8i load_frag_b64(const signed char* base, int h) {
  const int2 p0 = *(const int2*)(base + 0  + h * 8);
  const int2 p1 = *(const int2*)(base + 16 + h * 8);
  const int2 p2 = *(const int2*)(base + 32 + h * 8);
  const int2 p3 = *(const int2*)(base + 48 + h * 8);
  v8i f;
  f[0] = p0.x; f[1] = p0.y; f[2] = p1.x; f[3] = p1.y;
  f[4] = p2.x; f[5] = p2.y; f[6] = p3.x; f[7] = p3.y;
  return f;
}

// Issue one K-stage of async tile loads (A: 2 x b128/thread, B: 4 x b128/thread).
__device__ __forceinline__ void issue_stage_async(const signed char* __restrict__ xq,
                                                  const signed char* __restrict__ wq,
                                                  int m0, int n0, int kt,
                                                  unsigned asBase, unsigned bsBase, int t) {
#pragma unroll
  for (int i = 0; i < 2; ++i) {
    const int idx = t + i * 256, r = idx >> 3, c = (idx & 7) << 4;
    async_copy_b128(asBase + (unsigned)(r * LDSB + c),
                    xq + (size_t)(m0 + r) * K_DIM + kt + c);
  }
#pragma unroll
  for (int i = 0; i < 4; ++i) {
    const int idx = t + i * 256, r = idx >> 3, c = (idx & 7) << 4;
    async_copy_b128(bsBase + (unsigned)(r * LDSB + c),
                    wq + (size_t)(n0 + r) * K_DIM + kt + c);
  }
}

// ---------------- Kernel 3: int8 GEMM with WMMA IU8 + fused dequant epilogue ----------
__global__ __launch_bounds__(256) void w8a8_gemm_kernel(
    const signed char* __restrict__ xq, const float* __restrict__ xs,
    const signed char* __restrict__ wq, const float* __restrict__ wscale,
    const _Float16* __restrict__ bias, _Float16* __restrict__ out) {
  __shared__ signed char As[2][BM * LDSB];   // 18,432 B
  __shared__ signed char Bs[2][BN * LDSB];   // 36,864 B  (total 55,296 B)

  const int t = threadIdx.x;
  const int lane = t & 31;
  const int wv = t >> 5;        // wave 0..7
  const int wm = wv >> 2;       // 0..1 -> M offset wm*32
  const int wn = wv & 3;        // 0..3 -> N offset wn*32
  const int h  = lane >> 4;     // half-wave selector for K groups
  const int lr = lane & 15;     // row (A) / col (B/C) within 16
  const int m0 = blockIdx.y * BM;
  const int n0 = blockIdx.x * BN;

  // LDS byte offsets of the two double-buffer stages (low 32 bits of flat LDS addr).
  const unsigned asBase0 = (unsigned)(unsigned long long)(const void*)&As[0][0];
  const unsigned asBase1 = (unsigned)(unsigned long long)(const void*)&As[1][0];
  const unsigned bsBase0 = (unsigned)(unsigned long long)(const void*)&Bs[0][0];
  const unsigned bsBase1 = (unsigned)(unsigned long long)(const void*)&Bs[1][0];

  v8i acc[2][2];
#pragma unroll
  for (int i = 0; i < 2; ++i)
#pragma unroll
    for (int j = 0; j < 2; ++j) acc[i][j] = (v8i){0, 0, 0, 0, 0, 0, 0, 0};

  // ---- prologue: DMA stage 0 into buffer 0 ----
  issue_stage_async(xq, wq, m0, n0, /*kt=*/0, asBase0, bsBase0, t);
  wait_async0();
  __syncthreads();

  const int NSTAGE = K_DIM / BK;  // 32
  for (int s = 0; s < NSTAGE; ++s) {
    const int cur = s & 1;

    // Kick off DMA of stage s+1 into the other buffer (its last readers were
    // fenced by the end-of-stage s-1 barrier), overlapping the WMMAs below.
    if (s + 1 < NSTAGE) {
      issue_stage_async(xq, wq, m0, n0, (s + 1) * BK,
                        cur ? asBase0 : asBase1, cur ? bsBase0 : bsBase1, t);
    }
    // L2 prefetch of stage s+2 (one cacheline per tile row).
    if (s + 2 < NSTAGE) {
      const int kt2 = (s + 2) * BK;
      __builtin_prefetch(xq + (size_t)(m0 + (t & 63)) * K_DIM + kt2, 0, 1);
      __builtin_prefetch(wq + (size_t)(n0 + (t & 127)) * K_DIM + kt2, 0, 1);
    }

    const signed char* Ab = As[cur];
    const signed char* Bb = Bs[cur];
#pragma unroll
    for (int ks = 0; ks < 2; ++ks) {
      v8i fa[2], fb[2];
#pragma unroll
      for (int tm = 0; tm < 2; ++tm)
        fa[tm] = load_frag_b64(Ab + (wm * 32 + tm * 16 + lr) * LDSB + ks * 64, h);
#pragma unroll
      for (int tn = 0; tn < 2; ++tn)
        fb[tn] = load_frag_b64(Bb + (wn * 32 + tn * 16 + lr) * LDSB + ks * 64, h);
#pragma unroll
      for (int tm = 0; tm < 2; ++tm)
#pragma unroll
        for (int tn = 0; tn < 2; ++tn)
          acc[tm][tn] = __builtin_amdgcn_wmma_i32_16x16x64_iu8(
              /*sgn_a=*/true, fa[tm], /*sgn_b=*/true, fb[tn], acc[tm][tn],
              /*reuse_a=*/false, /*reuse_b=*/false);
    }

    // Publish next stage: my DMAs done, then workgroup-wide fence (single barrier/stage).
    wait_async0();
    __syncthreads();
  }

  // ---- epilogue: dequant + bias, fp16 store ----
  // C layout: VGPR r -> M = r + 8*h (lanes 16-31 are M+8), N = lane&15.
#pragma unroll
  for (int tm = 0; tm < 2; ++tm) {
    const int mbase = m0 + wm * 32 + tm * 16 + h * 8;  // 8 consecutive token rows
    const float4 s0 = *(const float4*)(xs + mbase);
    const float4 s1 = *(const float4*)(xs + mbase + 4);
    const float sc[8] = {s0.x, s0.y, s0.z, s0.w, s1.x, s1.y, s1.z, s1.w};
#pragma unroll
    for (int tn = 0; tn < 2; ++tn) {
      const int o = n0 + wn * 32 + tn * 16 + lr;
      const float wsc = wscale[o];
      const float bo = (float)bias[o];
#pragma unroll
      for (int r = 0; r < 8; ++r) {
        const float vv = (float)acc[tm][tn][r] * sc[r] * wsc + bo;
        out[(size_t)(mbase + r) * N_DIM + o] = (_Float16)vv;
      }
    }
  }
}

// ---------------- Host launcher ----------------
extern "C" void kernel_launch(void* const* d_in, const int* in_sizes, int n_in,
                              void* d_out, int out_size, void* d_ws, size_t ws_size,
                              hipStream_t stream) {
  (void)in_sizes; (void)n_in; (void)out_size; (void)ws_size;
  const _Float16* x    = (const _Float16*)d_in[0];   // fp16 activations [B,S,IN]
  const int*      w    = (const int*)d_in[1];        // int8 weights delivered as int32
  const float*    wsc  = (const float*)d_in[2];      // per-channel scale [OUT]
  const _Float16* bs   = (const _Float16*)d_in[3];   // fp16 bias [OUT]
  _Float16*       outp = (_Float16*)d_out;           // fp16 output [B,S,OUT]

  // workspace layout (all 16B aligned)
  signed char* xq = (signed char*)d_ws;                          // 32 MB
  signed char* wq = xq + (size_t)M_TOK * K_DIM;                  // 48 MB
  float*       xs = (float*)(wq + (size_t)N_DIM * K_DIM);        // 32 KB

  quant_x_kernel<<<M_TOK, 256, 0, stream>>>(x, xq, xs);
  pack_w_kernel<<<((size_t)N_DIM * K_DIM) / (4 * 256), 256, 0, stream>>>(w, (unsigned int*)wq);
  w8a8_gemm_kernel<<<dim3(N_DIM / BN, M_TOK / BM), 256, 0, stream>>>(xq, xs, wq, wsc, bs, outp);
}
